// ReservoirLayer_90494960927022
// MI455X (gfx1250) — matrix operand order repaired and checked by reading, not compiled, not run on hardware
//
#include <hip/hip_runtime.h>
#include <math.h>

// ---------------------------------------------------------------------------
// ESN leaky-integrator scan for MI455X (gfx1250), bf16 WMMA + f32 state.
//   Phase 1: u = x @ W_in^T + b   (written into d_out, in place of output)
//   Phase 2: persistent scan, 32 WGs (2 batch-tiles x 16 column-slices),
//            per-step inter-WG exchange via global buffer + counter barrier,
//            state re-broadcast into LDS with global_load_async_to_lds_b128.
// ---------------------------------------------------------------------------

#define GAMMA_F 0.95f

#define B_    32
#define T_    512
#define F_    128
#define R_    2048
#define MROWS (B_ * T_)            // 16384 rows of the input GEMM

// LDS layouts (bytes / element strides chosen to spread 16 lanes over 64 banks)
#define SBF_STRIDE  2056           // ushorts per state row: 2048 + 8 pad (4112 B)
#define SF32_STRIDE 132            // floats per f32-slice row: 128 + 4 pad

typedef __attribute__((ext_vector_type(16))) __bf16 v16bf;
typedef __attribute__((ext_vector_type(8)))  float  v8f;

union ABFrag { uint4 q[2]; v16bf v; };

__device__ __forceinline__ unsigned short f2bf(float f) {
  unsigned int u = __float_as_uint(f);
  u += 0x7FFFu + ((u >> 16) & 1u);           // round-to-nearest-even
  return (unsigned short)(u >> 16);
}

// ---------------------------------------------------------------------------
// f32 -> bf16 row-major conversion (grid-stride)
// ---------------------------------------------------------------------------
__global__ void esn_cvt_bf16(const float* __restrict__ src,
                             unsigned short* __restrict__ dst, int n) {
  for (int i = blockIdx.x * blockDim.x + threadIdx.x; i < n;
       i += gridDim.x * blockDim.x)
    dst[i] = f2bf(src[i]);
}

__global__ void esn_zero_cnt(unsigned int* c) {
  if (threadIdx.x < 8) c[threadIdx.x] = 0u;
}

// ---------------------------------------------------------------------------
// Phase 1: u[m][n] = sum_k x[m][k] * W_in[n][k] + b[n]
// One wave per 16x16 output tile, K=128 (4 x wmma_f32_16x16x32_bf16).
// A and B fragments load directly from row-major bf16 with 2 x b128 each.
// ---------------------------------------------------------------------------
__global__ void __launch_bounds__(256)
esn_input_proj(const unsigned short* __restrict__ xbf,   // [16384][128] bf16
               const unsigned short* __restrict__ wbf,   // [2048][128]  bf16
               const float* __restrict__ bias,           // [2048]
               float* __restrict__ u)                    // [16384][2048] f32
{
  const int lane  = threadIdx.x & 31;
  const int wv    = threadIdx.x >> 5;                    // 0..7
  const int mtile = blockIdx.x >> 4;                     // 0..1023
  const int ntile = ((blockIdx.x & 15) << 3) + wv;       // 0..127
  const int lm    = lane & 15;
  const int hf    = lane >> 4;                           // half-wave select

  const int mrow = mtile * 16 + lm;                      // A row (x row)
  const int ncol = ntile * 16 + lm;                      // B row (W_in row) / C col

  const char* arow = (const char*)(xbf + (size_t)mrow * F_);
  const char* brow = (const char*)(wbf + (size_t)ncol * F_);

  const float bv = bias[ncol];
  v8f c;
#pragma unroll
  for (int i = 0; i < 8; ++i) c[i] = bv;

#pragma unroll
  for (int kt = 0; kt < F_ / 32; ++kt) {
    ABFrag a, b;
    a.q[0] = *(const uint4*)(arow + kt * 64 + hf * 16);        // K = hf*8 .. +7
    a.q[1] = *(const uint4*)(arow + kt * 64 + 32 + hf * 16);   // K = 16+hf*8 ..
    b.q[0] = *(const uint4*)(brow + kt * 64 + hf * 16);        // K = hf*8 .. +7
    b.q[1] = *(const uint4*)(brow + kt * 64 + 32 + hf * 16);   // K = 16+hf*8 ..
    c = __builtin_amdgcn_wmma_f32_16x16x32_bf16(
            false, a.v, false, b.v, (short)0, c, false, false);
  }

  float* outp = u + (size_t)(mtile * 16) * R_ + ncol;
#pragma unroll
  for (int i = 0; i < 8; ++i) {
    const int m = i + hf * 8;
    outp[(size_t)m * R_] = c[i];
  }
}

// ---------------------------------------------------------------------------
// Phase 2: persistent scan kernel.
// Grid: 32 WGs = 2 batch-tiles x 16 column-slices, 256 threads (8 waves).
// Wave wv owns N-tile wv of its 128-column slice; K=2048 -> 64 WMMAs/step.
// A fragments: ds_load_b128 from the LDS bf16 state image.
// B fragments: b128 streams from L2-resident bf16 W_rec.
// Per-step exchange: write own f32 result to d_out + bf16 slice to global
// double buffer, release/acquire counter barrier over the 16 WGs of the
// batch-tile, then async-copy the full bf16 state back into LDS.
// ---------------------------------------------------------------------------
__global__ void __launch_bounds__(256, 1)
esn_scan(const unsigned short* __restrict__ wrec,   // [2048][2048] bf16 row-major
         const float* __restrict__ state0,          // [2048]
         float* __restrict__ out,                   // [32][512][2048]: u in, s out
         unsigned short* __restrict__ sx,           // [2 buf][2 bt][16][2048] bf16
         unsigned int* __restrict__ cnt)            // [2] arrival counters
{
  extern __shared__ char smem[];
  unsigned short* s_bf  = (unsigned short*)smem;                    // [16][2056]
  float*          s_f32 = (float*)(smem + 16 * SBF_STRIDE * 2);     // [16][132]

  const int bt      = blockIdx.x >> 4;       // batch tile 0..1
  const int cs      = blockIdx.x & 15;       // column slice 0..15
  const int colbase = cs * 128;
  const int tid     = threadIdx.x;
  const int lane    = tid & 31;
  const int wv      = tid >> 5;              // wave -> N-tile within slice
  const int lm      = lane & 15;
  const int hf      = lane >> 4;

  // ---- init: broadcast state0 into LDS (bf16 full image + f32 own slice)
  for (int i = tid; i < 16 * R_; i += 256) {
    const int m = i >> 11, k = i & (R_ - 1);
    s_bf[m * SBF_STRIDE + k] = f2bf(state0[k]);
  }
  for (int i = tid; i < 16 * 128; i += 256) {
    const int m = i >> 7, j = i & 127;
    s_f32[m * SF32_STRIDE + j] = state0[colbase + j];
  }
  __syncthreads();

  const int   ncol = colbase + wv * 16 + lm;                       // global col
  const int   jloc = wv * 16 + lm;                                 // col in slice
  const char* brow = (const char*)(wrec + (size_t)ncol * R_);
  const char* arow = (const char*)s_bf + (size_t)lm * SBF_STRIDE * 2;
  unsigned int* mycnt = cnt + bt;

  for (int t = 0; t < T_; ++t) {
    // C init = u[b][t][ncol]  (b = bt*16 + m), read from d_out in place
    float* uptr = out + ((size_t)(bt * 16) * T_ + t) * R_ + ncol;
    v8f c;
#pragma unroll
    for (int i = 0; i < 8; ++i) {
      const int m = i + hf * 8;
      c[i] = uptr[(size_t)m * (T_ * R_)];
    }

    // y += s @ W_rec^T over K=2048
#pragma unroll 4
    for (int kt = 0; kt < R_ / 32; ++kt) {
      ABFrag a, b;
      a.q[0] = *(const uint4*)(arow + kt * 64 + hf * 16);
      a.q[1] = *(const uint4*)(arow + kt * 64 + 32 + hf * 16);
      b.q[0] = *(const uint4*)(brow + kt * 64 + hf * 16);
      b.q[1] = *(const uint4*)(brow + kt * 64 + 32 + hf * 16);
      c = __builtin_amdgcn_wmma_f32_16x16x32_bf16(
              false, a.v, false, b.v, (short)0, c, false, false);
    }

    // leaky update in f32, scatter results
    unsigned short* sxrow =
        sx + ((size_t)((t & 1) * 2 + bt) * 16) * R_ + ncol;
#pragma unroll
    for (int i = 0; i < 8; ++i) {
      const int m = i + hf * 8;
      const float sold = s_f32[m * SF32_STRIDE + jloc];
      const float snew = (1.0f - GAMMA_F) * sold + GAMMA_F * tanhf(c[i]);
      s_f32[m * SF32_STRIDE + jloc] = snew;
      uptr[(size_t)m * (T_ * R_)]   = snew;           // ys[b][t][ncol]
      sxrow[(size_t)m * R_]         = f2bf(snew);     // bf16 exchange slice
    }

    // ---- inter-WG barrier over the 16 WGs of this batch tile
    __threadfence();                                   // agent-scope release
    __syncthreads();
    if (tid == 0)
      __hip_atomic_fetch_add(mycnt, 1u, __ATOMIC_RELEASE,
                             __HIP_MEMORY_SCOPE_AGENT);
    const unsigned int target = 16u * (unsigned)(t + 1);
    while (__hip_atomic_load(mycnt, __ATOMIC_ACQUIRE,
                             __HIP_MEMORY_SCOPE_AGENT) < target) {
      __builtin_amdgcn_s_sleep(1);
    }

    // ---- pull the full new bf16 state into LDS via async copy (CDNA5 path)
    if (t + 1 < T_) {
      const char* sxt =
          (const char*)(sx + (size_t)((t & 1) * 2 + bt) * 16 * R_);
#pragma unroll
      for (int i = 0; i < 16; ++i) {                   // 16 rows x 4096 B
        const unsigned lds_off = (unsigned)(i * SBF_STRIDE * 2 + tid * 16);
        const void* gp = (const void*)(sxt + (size_t)i * R_ * 2 + tid * 16);
        asm volatile("global_load_async_to_lds_b128 %0, %1, off"
                     :: "v"(lds_off), "v"(gp) : "memory");
      }
      asm volatile("s_wait_asynccnt 0x0" ::: "memory");
      __syncthreads();
    }
  }
}

// ---------------------------------------------------------------------------
// Launcher
// ---------------------------------------------------------------------------
extern "C" void kernel_launch(void* const* d_in, const int* in_sizes, int n_in,
                              void* d_out, int out_size, void* d_ws, size_t ws_size,
                              hipStream_t stream) {
  (void)in_sizes; (void)n_in; (void)out_size; (void)ws_size;

  const float* x      = (const float*)d_in[0];   // [32][512][128]
  const float* W_in   = (const float*)d_in[1];   // [2048][128]
  const float* W_rec  = (const float*)d_in[2];   // [2048][2048]
  const float* bias   = (const float*)d_in[3];   // [2048]
  const float* state0 = (const float*)d_in[4];   // [2048]
  float* out = (float*)d_out;                    // [32][512][2048]

  char* ws = (char*)d_ws;
  const size_t xbf_off   = 0;                                   //  4 MiB
  const size_t winbf_off = xbf_off   + (size_t)MROWS * F_ * 2;  // 512 KiB
  const size_t wrec_off  = winbf_off + (size_t)R_ * F_ * 2;     //  8 MiB
  const size_t sx_off    = wrec_off  + (size_t)R_ * R_ * 2;     // 256 KiB
  const size_t cnt_off   = sx_off    + (size_t)2 * 2 * 16 * R_ * 2;

  unsigned short* xbf    = (unsigned short*)(ws + xbf_off);
  unsigned short* winbf  = (unsigned short*)(ws + winbf_off);
  unsigned short* wrecbf = (unsigned short*)(ws + wrec_off);
  unsigned short* sx     = (unsigned short*)(ws + sx_off);
  unsigned int*   cnt    = (unsigned int*)(ws + cnt_off);

  // reset arrival counters (ws is not re-zeroed by the harness)
  esn_zero_cnt<<<1, 32, 0, stream>>>(cnt);

  // f32 -> bf16 staging (rewritten every call; deterministic)
  esn_cvt_bf16<<<1024, 256, 0, stream>>>(x,     xbf,    MROWS * F_);
  esn_cvt_bf16<<<256,  256, 0, stream>>>(W_in,  winbf,  R_ * F_);
  esn_cvt_bf16<<<2048, 256, 0, stream>>>(W_rec, wrecbf, R_ * R_);

  // Phase 1: u into d_out. 1024 M-tiles x 128 N-tiles, 8 wave-tiles/block.
  esn_input_proj<<<16384, 256, 0, stream>>>(xbf, winbf, bias, out);

  // Phase 2: persistent scan, 32 co-resident WGs.
  const size_t lds_bytes = (size_t)16 * SBF_STRIDE * 2   // bf16 state image
                         + (size_t)16 * SF32_STRIDE * 4; // f32 own slice
  esn_scan<<<32, 256, lds_bytes, stream>>>(wrecbf, state0, out, sx, cnt);
}
